// CausalSelfAttention_4114578669975
// MI455X (gfx1250) — matrix operand order repaired and checked by reading, not compiled, not run on hardware
//
#include <hip/hip_runtime.h>
#include <hip/hip_bf16.h>

// ---------- constants ----------
#define BATCH 4
#define SEQ   2048
#define EMB   768
#define HEADS 12
#define HDIM  64
#define MROWS (BATCH * SEQ)        // 8192
#define NQKV  (3 * EMB)            // 2304

typedef _Float16 v16h __attribute__((ext_vector_type(16)));
typedef _Float16 v8h  __attribute__((ext_vector_type(8)));
typedef float    v8f  __attribute__((ext_vector_type(8)));

static __device__ __forceinline__ v16h cat8(v8h lo, v8h hi) {
    return __builtin_shufflevector(lo, hi, 0,1,2,3,4,5,6,7,8,9,10,11,12,13,14,15);
}

static __device__ __forceinline__ v8f wmma_f16(v16h a, v16h b, v8f c) {
    // D = A(16x32 f16) x B(32x16 f16) + C(16x16 f32)
    return __builtin_amdgcn_wmma_f32_16x16x32_f16(false, a, false, b, (short)0, c, false, false);
}

// workgroup-relative LDS byte offset of a generic pointer to __shared__
static __device__ __forceinline__ unsigned lds_addr_of(const void* p) {
    return (unsigned)(unsigned long long)(const __attribute__((address_space(3))) char*)p;
}

static __device__ __forceinline__ void async_copy_b128(unsigned lds_byte, unsigned long long gaddr) {
    // LDS[lds_byte .. +15] = MEM[gaddr .. +15], tracked by ASYNCcnt
    asm volatile("global_load_async_to_lds_b128 %0, %1, off" :: "v"(lds_byte), "v"(gaddr) : "memory");
}

// ---------- small conversion kernels ----------
__global__ void k_f32_to_f16(const float* __restrict__ in, _Float16* __restrict__ out, int n) {
    int i = blockIdx.x * blockDim.x + threadIdx.x;
    int stride = gridDim.x * blockDim.x;
    for (; i < n; i += stride) out[i] = (_Float16)in[i];
}

// out[c*rows + r] = (f16) in[r*cols + c]   (store W^T so B-fragments are contiguous in K)
__global__ void k_transpose_f16(const float* __restrict__ in, _Float16* __restrict__ out,
                                int rows, int cols) {
    int i = blockIdx.x * blockDim.x + threadIdx.x;
    if (i >= rows * cols) return;
    int r = i / cols, c = i % cols;
    out[(size_t)c * rows + r] = (_Float16)in[i];
}

// ---------- QKV projection: qkv = x @ Wqkv + bqkv, scattered to Q/K [B,H,T,D], V^T [B,H,D,T] ----------
__global__ __launch_bounds__(128) void k_qkv_gemm(const _Float16* __restrict__ xh,
                                                  const _Float16* __restrict__ wt,   // [2304 x 768] = Wqkv^T
                                                  const float* __restrict__ bias,
                                                  _Float16* __restrict__ q,
                                                  _Float16* __restrict__ kk,
                                                  _Float16* __restrict__ vt) {
    const int lane = threadIdx.x;
    const int ln = lane & 15, hf = lane >> 4;
    const int ngrp  = blockIdx.x * 4 + threadIdx.y;     // 0..35  (each covers 4 N-tiles)
    const int mtile = blockIdx.y;                        // 0..511
    const int nt0   = ngrp * 4;

    const _Float16* arow = xh + (size_t)(mtile * 16 + ln) * EMB;
    v8f acc[4] = { {}, {}, {}, {} };

    for (int k0 = 0; k0 < EMB; k0 += 32) {
        v8h alo = *(const v8h*)(arow + k0 + hf * 8);
        v8h ahi = *(const v8h*)(arow + k0 + 16 + hf * 8);
        v16h a = cat8(alo, ahi);
#pragma unroll
        for (int u = 0; u < 4; ++u) {
            const _Float16* brow = wt + (size_t)((nt0 + u) * 16 + ln) * EMB;
            v16h b = cat8(*(const v8h*)(brow + k0 + hf * 8), *(const v8h*)(brow + k0 + 16 + hf * 8));
            acc[u] = wmma_f16(a, b, acc[u]);
        }
    }

#pragma unroll
    for (int u = 0; u < 4; ++u) {
        int n = (nt0 + u) * 16 + ln;          // global output column in [0,2304)
        float bv = bias[n];
        int sec = n / EMB;                    // 0=q 1=k 2=v
        int cc  = n % EMB;
        int h = cc >> 6, d = cc & 63;
#pragma unroll
        for (int j = 0; j < 8; ++j) {
            int row = mtile * 16 + j + 8 * hf;
            int bidx = row >> 11, t = row & (SEQ - 1);
            _Float16 hv = (_Float16)(acc[u][j] + bv);
            if (sec == 0)
                q[((size_t)(bidx * HEADS + h) * SEQ + t) * HDIM + d] = hv;
            else if (sec == 1)
                kk[((size_t)(bidx * HEADS + h) * SEQ + t) * HDIM + d] = hv;
            else
                vt[((size_t)(bidx * HEADS + h) * HDIM + d) * SEQ + t] = hv;
        }
    }
}

// ---------- flash attention: one wave per (b,h,32-row q tile), K/V staged via async-to-LDS ----------
__global__ __launch_bounds__(128) void k_attn(const _Float16* __restrict__ q,
                                              const _Float16* __restrict__ kk,
                                              const _Float16* __restrict__ vt,
                                              _Float16* __restrict__ aout) {
    __shared__ _Float16 kTile[4][32 * 64];        // 4KB/wave: local kt row-major [32][64]
    __shared__ _Float16 vTile[4][64 * 32];        // 4KB/wave: [d][t_local]
    __shared__ _Float16 pTile[4][2][16 * 40];     // P round-trip, padded stride 40

    const int lane = threadIdx.x;
    const int ln = lane & 15, hf = lane >> 4;
    const int w = threadIdx.y;
    const int idx = blockIdx.x * 4 + w;           // 0..3071
    const int bh = idx >> 6;                      // 0..47
    const int qbase = (idx & 63) * 32;

    const _Float16* qb = q  + (size_t)bh * SEQ * HDIM;
    const _Float16* kb = kk + (size_t)bh * SEQ * HDIM;
    const _Float16* vb = vt + (size_t)bh * HDIM * SEQ;

    // Preload A-fragments of Q: two 16-row tiles, K-dim = d in two 32-wide steps
    v16h qa[2][2];
#pragma unroll
    for (int rt = 0; rt < 2; ++rt)
#pragma unroll
        for (int s = 0; s < 2; ++s) {
            const _Float16* r = qb + (size_t)(qbase + rt * 16 + ln) * HDIM + s * 32;
            qa[rt][s] = cat8(*(const v8h*)(r + hf * 8), *(const v8h*)(r + 16 + hf * 8));
        }

    float mrow[2][8], lrow[2][8];
    v8f acc[2][4] = { { {}, {}, {}, {} }, { {}, {}, {}, {} } };
#pragma unroll
    for (int rt = 0; rt < 2; ++rt)
#pragma unroll
        for (int j = 0; j < 8; ++j) { mrow[rt][j] = -3.0e38f; lrow[rt][j] = 0.0f; }

    const unsigned kl = lds_addr_of(&kTile[w][0]);
    const unsigned vl = lds_addr_of(&vTile[w][0]);
    const unsigned long long kgb = (unsigned long long)kb;
    const unsigned long long vgb = (unsigned long long)vb;

    const int nk = (qbase >> 5) + 1;              // causal: kt0 = 0..qbase

    for (int it = 0; it < nk; ++it) {
        const int kt0 = it * 32;

        // prior-iteration LDS reads must land before async writes overwrite the tiles
        asm volatile("s_wait_dscnt 0" ::: "memory");
        // K tile: contiguous 4KB  [kt0..kt0+31][0..63]
#pragma unroll
        for (int c8 = 0; c8 < 8; ++c8) {
            int c = c8 * 32 + lane;
            async_copy_b128(kl + c * 16, kgb + (unsigned long long)kt0 * 128 + (unsigned)(c * 16));
        }
        // V tile: 64 rows of 64B  [d][kt0..kt0+31]
#pragma unroll
        for (int c8 = 0; c8 < 8; ++c8) {
            int c = c8 * 32 + lane;
            int d = c >> 2, sub = c & 3;
            async_copy_b128(vl + c * 16,
                            vgb + (unsigned long long)d * (SEQ * 2) + (unsigned)(kt0 * 2 + sub * 16));
        }
        asm volatile("s_wait_asynccnt 0" ::: "memory");

        // K^T B-fragments from LDS: col n = local kt, K-dim = d
        v16h kf[2][2];
#pragma unroll
        for (int nsub = 0; nsub < 2; ++nsub)
#pragma unroll
            for (int s = 0; s < 2; ++s)
                kf[nsub][s] = cat8(*(const v8h*)&kTile[w][(nsub * 16 + ln) * 64 + s * 32 + hf * 8], *(const v8h*)&kTile[w][(nsub * 16 + ln) * 64 + s * 32 + 16 + hf * 8]);

        // S = Q x K^T : 2 row-tiles x 2 col-subtiles
        v8f sc[2][2] = { { {}, {} }, { {}, {} } };
#pragma unroll
        for (int rt = 0; rt < 2; ++rt)
#pragma unroll
            for (int nsub = 0; nsub < 2; ++nsub) {
                sc[rt][nsub] = wmma_f16(qa[rt][0], kf[nsub][0], sc[rt][nsub]);
                sc[rt][nsub] = wmma_f16(qa[rt][1], kf[nsub][1], sc[rt][nsub]);
            }

        const bool domask = (kt0 + 31 > qbase);   // only the diagonal block
#pragma unroll
        for (int rt = 0; rt < 2; ++rt) {
            _Float16* sm = &pTile[w][rt][0];
#pragma unroll
            for (int j = 0; j < 8; ++j) {
                float x0 = sc[rt][0][j] * 0.125f;     // 1/sqrt(64)
                float x1 = sc[rt][1][j] * 0.125f;
                if (domask) {
                    int row = qbase + rt * 16 + j + 8 * hf;
                    if (kt0 + ln > row)      x0 = -3.0e38f;
                    if (kt0 + 16 + ln > row) x1 = -3.0e38f;
                }
                float rm = fmaxf(x0, x1);
                rm = fmaxf(rm, __shfl_xor(rm, 1));
                rm = fmaxf(rm, __shfl_xor(rm, 2));
                rm = fmaxf(rm, __shfl_xor(rm, 4));
                rm = fmaxf(rm, __shfl_xor(rm, 8));
                float mn = fmaxf(mrow[rt][j], rm);
                float cs = __expf(mrow[rt][j] - mn);
                float p0 = __expf(x0 - mn);
                float p1 = __expf(x1 - mn);
                float rs = p0 + p1;
                rs += __shfl_xor(rs, 1);
                rs += __shfl_xor(rs, 2);
                rs += __shfl_xor(rs, 4);
                rs += __shfl_xor(rs, 8);
                lrow[rt][j] = lrow[rt][j] * cs + rs;
                mrow[rt][j] = mn;
#pragma unroll
                for (int t = 0; t < 4; ++t) acc[rt][t][j] *= cs;
                int m = j + 8 * hf;
                sm[m * 40 + ln]      = (_Float16)p0;
                sm[m * 40 + 16 + ln] = (_Float16)p1;
            }
        }
        asm volatile("s_wait_dscnt 0" ::: "memory");

        // P A-fragments and V B-fragments from LDS
        v16h pa[2];
#pragma unroll
        for (int rt = 0; rt < 2; ++rt) {
            const _Float16* sm = &pTile[w][rt][0];
            v8h plo = *(const v8h*)(sm + ln * 40 + hf * 8);
            v8h phi = *(const v8h*)(sm + ln * 40 + 16 + hf * 8);
            pa[rt] = cat8(plo, phi);
        }
        v16h vf[4];
#pragma unroll
        for (int t = 0; t < 4; ++t)
            vf[t] = cat8(*(const v8h*)&vTile[w][(t * 16 + ln) * 32 + hf * 8], *(const v8h*)&vTile[w][(t * 16 + ln) * 32 + 16 + hf * 8]);
#pragma unroll
        for (int rt = 0; rt < 2; ++rt)
#pragma unroll
            for (int t = 0; t < 4; ++t)
                acc[rt][t] = wmma_f16(pa[rt], vf[t], acc[rt][t]);
    }

    // normalize + write attention output as f16 [B,T,C]
    const int b = bh / HEADS, h = bh % HEADS;
#pragma unroll
    for (int rt = 0; rt < 2; ++rt)
#pragma unroll
        for (int j = 0; j < 8; ++j) {
            float inv = 1.0f / lrow[rt][j];
            int trow = qbase + rt * 16 + j + 8 * hf;
            _Float16* orow = aout + ((size_t)(b * SEQ + trow)) * EMB + h * HDIM;
#pragma unroll
            for (int t = 0; t < 4; ++t) orow[t * 16 + ln] = (_Float16)(acc[rt][t][j] * inv);
        }
}

// ---------- output projection: out = aout @ Wout + bout (fp32 result) ----------
__global__ __launch_bounds__(128) void k_out_gemm(const _Float16* __restrict__ ah,
                                                  const _Float16* __restrict__ wt,   // [768 x 768] = Wout^T
                                                  const float* __restrict__ bias,
                                                  float* __restrict__ out) {
    const int lane = threadIdx.x;
    const int ln = lane & 15, hf = lane >> 4;
    const int ngrp  = blockIdx.x * 4 + threadIdx.y;     // 0..11 (each covers 4 N-tiles)
    const int mtile = blockIdx.y;                        // 0..511
    const int nt0   = ngrp * 4;

    const _Float16* arow = ah + (size_t)(mtile * 16 + ln) * EMB;
    v8f acc[4] = { {}, {}, {}, {} };

    for (int k0 = 0; k0 < EMB; k0 += 32) {
        v8h alo = *(const v8h*)(arow + k0 + hf * 8);
        v8h ahi = *(const v8h*)(arow + k0 + 16 + hf * 8);
        v16h a = cat8(alo, ahi);
#pragma unroll
        for (int u = 0; u < 4; ++u) {
            const _Float16* brow = wt + (size_t)((nt0 + u) * 16 + ln) * EMB;
            v16h b = cat8(*(const v8h*)(brow + k0 + hf * 8), *(const v8h*)(brow + k0 + 16 + hf * 8));
            acc[u] = wmma_f16(a, b, acc[u]);
        }
    }

#pragma unroll
    for (int u = 0; u < 4; ++u) {
        int n = (nt0 + u) * 16 + ln;
        float bv = bias[n];
#pragma unroll
        for (int j = 0; j < 8; ++j) {
            int row = mtile * 16 + j + 8 * hf;
            out[(size_t)row * EMB + n] = acc[u][j] + bv;
        }
    }
}

// ---------- launcher ----------
extern "C" void kernel_launch(void* const* d_in, const int* in_sizes, int n_in,
                              void* d_out, int out_size, void* d_ws, size_t ws_size,
                              hipStream_t stream) {
    const float* x    = (const float*)d_in[0];   // [4,2048,768]
    const float* Wqkv = (const float*)d_in[1];   // [768,2304]
    const float* bqkv = (const float*)d_in[2];   // [2304]
    const float* Wout = (const float*)d_in[3];   // [768,768]
    const float* bout = (const float*)d_in[4];   // [768]
    float* outp = (float*)d_out;

    const size_t XH = (size_t)MROWS * EMB;       // 6,291,456 halves
    const size_t WQ = (size_t)EMB * NQKV;        // 1,769,472
    const size_t WO = (size_t)EMB * EMB;         //   589,824
    const size_t QS = (size_t)BATCH * HEADS * SEQ * HDIM;  // = XH

    char* p = (char*)d_ws;
    _Float16* xh  = (_Float16*)p;                 p += XH * 2;
    _Float16* wq  = (_Float16*)p;                 p += WQ * 2;
    _Float16* wo  = (_Float16*)p;                 p += WO * 2;
    _Float16* qb  = (_Float16*)p;                 p += QS * 2;
    _Float16* kb  = (_Float16*)p;                 p += QS * 2;
    _Float16* vtb = (_Float16*)p;                 p += QS * 2;
    _Float16* ao  = (_Float16*)p;                 p += QS * 2;

    k_f32_to_f16<<<4096, 256, 0, stream>>>(x, xh, (int)XH);
    k_transpose_f16<<<(unsigned)((WQ + 255) / 256), 256, 0, stream>>>(Wqkv, wq, EMB, NQKV);
    k_transpose_f16<<<(unsigned)((WO + 255) / 256), 256, 0, stream>>>(Wout, wo, EMB, EMB);

    k_qkv_gemm<<<dim3(9, 512), dim3(32, 4), 0, stream>>>(xh, wq, bqkv, qb, kb, vtb);
    k_attn<<<dim3(768), dim3(32, 4), 0, stream>>>(qb, kb, vtb, ao);
    k_out_gemm<<<dim3(3, 512), dim3(32, 4), 0, stream>>>(ao, wo, bout, outp);
}
